// dkt_78580721648117
// MI455X (gfx1250) — matrix operand order repaired and checked by reading, not compiled, not run on hardware
//
#include <hip/hip_runtime.h>

// DKT GRU scan for MI455X (gfx1250, wave32, WMMA).
//
// Per step t:
//   tmp[256 x 7168] = [ h @ w1 | h @ w_hh | x_t @ w_ih ]   (bf16 WMMA, fp32 acc)
//   epilogue: z = relu(tmp[:, :1024] + b1)
//             prob[b] = sigmoid(dot(z[b], wo[:, pid[b]]) + bo[pid[b]])  (skip full z@wo)
//             GRU gates from tmp[:, 1024:7168] -> h_new (fp32)
//             pack h_new -> bf16 A-fragments; gather+pack x_{t+1}
//
// GEMM: 4 M-tiles x 4 N-tiles per wave (128 acc VGPRs), register double-buffered
// (ping/pong) so WMMAs overlap the next K-tile's fragment loads instead of
// stalling on s_wait_loadcnt 0.

typedef __attribute__((ext_vector_type(16))) __bf16 v16bf;
typedef __attribute__((ext_vector_type(8)))  float  v8f;

#define BDIM   256
#define TSTEPS 200
#define DIMK   1024
#define PDIM   12000
#define NCOLS  7168              // [w1:1024 | w_hh:3072 | w_ih:3072]
#define NTILES (NCOLS / 16)      // 448
#define KTILES (DIMK / 32)       // 32
#define MTILES (BDIM / 16)       // 16
#define TILE_US 512              // ushorts per 16x32 / 32x16 fragment tile
#define TSTRIDE ((size_t)KTILES * TILE_US)   // ushorts between tile rows

// ---------- helpers ----------

__device__ __forceinline__ unsigned short f2bf(float f) {
  union { float f; unsigned u; } x; x.f = f;
  unsigned u = x.u;
  return (unsigned short)((u + 0x7FFFu + ((u >> 16) & 1u)) >> 16);  // RNE
}

// A-matrix 16x32 bf16 fragment layout (ISA 7.12.2):
// lanes 0-15: V0..V3 K=0..7, V4..V7 K=16..23 ; lanes 16-31: K=8..15 / 24..31.
// Per-lane 16 bf16 contiguous (vgpr-major, 2 bf16 per vgpr).
// For kk0 % 4 == 0 (within one 8-block), kk0..kk0+3 are 4 contiguous ushorts.
__device__ __forceinline__ int a_frag_off(int m, int kk) {
  int hl   = (kk >> 3) & 1;
  int lane = m + (hl << 4);
  int v    = ((kk & 7) >> 1) + ((kk >> 4) << 2);
  return lane * 16 + v * 2 + (kk & 1);
}

__device__ __forceinline__ float sigmoidf(float x) {
  return 1.0f / (1.0f + __expf(-x));
}

union F4 { float4 v; float f[4]; };

// ---------- kernel 1: pack [w1|w_hh|w_ih] fp32 -> bf16 B-fragment order ----------
__global__ void __launch_bounds__(256)
dkt_pack_w(const float* __restrict__ w1, const float* __restrict__ w_hh,
           const float* __restrict__ w_ih, unsigned short* __restrict__ Bp) {
  int idx    = blockIdx.x * 256 + threadIdx.x;   // 448*32*512 elems
  int tile   = idx >> 9;
  int within = idx & 511;
  int lane   = within >> 4;
  int rest   = within & 15;
  int v      = rest >> 1;
  int by     = rest & 1;
  int ntile  = tile >> 5;
  int kt     = tile & 31;
  int nn     = lane & 15;
  int kk     = ((lane >> 4) << 3) + ((v & 3) << 1) + by + ((v >> 2) << 4);
  int k      = kt * 32 + kk;
  int col    = ntile * 16 + nn;
  float s;
  if (col < 1024)      s = w1  [(size_t)k * 1024 + col];
  else if (col < 4096) s = w_hh[(size_t)k * 3072 + (col - 1024)];
  else                 s = w_ih[(size_t)k * 3072 + (col - 4096)];
  Bp[idx] = f2bf(s);
}

// ---------- kernel 2: init h = 0, hA = 0, pack x for t = 0 ----------
__global__ void __launch_bounds__(256)
dkt_init(const int* __restrict__ prob_ids, const int* __restrict__ operate,
         const float* __restrict__ prob_emb, float* __restrict__ h,
         unsigned short* __restrict__ hA, unsigned short* __restrict__ xA) {
  int idx = blockIdx.x * 256 + threadIdx.x;   // 256*1024
  h[idx]  = 0.0f;
  hA[idx] = 0;
  int b = idx >> 10, j = idx & 1023;
  int pid = prob_ids[b];
  int op  = operate[b];
  float val = prob_emb[(size_t)(pid * 2 + op) * DIMK + j];
  xA[((b >> 4) * KTILES + (j >> 5)) * TILE_US + a_frag_off(b & 15, j & 31)] = f2bf(val);
}

// ---------- kernel 3: per-step GEMM  tmp = [h|x] @ Wcat ----------
// grid (14, 4), block 256 (8 waves). Wave: 4 M-tiles x 4 N-tiles.
__global__ void __launch_bounds__(256)
dkt_gemm(const unsigned short* __restrict__ hA, const unsigned short* __restrict__ xA,
         const unsigned short* __restrict__ Bp, float* __restrict__ tmp) {
  int lane  = threadIdx.x & 31;
  int wid   = threadIdx.x >> 5;
  int mtile = blockIdx.y * 4;
  int ntile = (blockIdx.x * 8 + wid) * 4;
  // ntiles [0,256) consume h (w1, w_hh); [256,448) consume x (w_ih)
  const unsigned short* Ab = (ntile < 256 ? hA : xA) + (size_t)mtile * TSTRIDE + lane * 16;
  const unsigned short* Bb = Bp + (size_t)ntile * TSTRIDE + lane * 16;

  v8f acc[4][4] = {};
  v16bf a0[4], b0[4], a1[4], b1[4];

#pragma unroll
  for (int i = 0; i < 4; ++i) {
    a0[i] = *(const v16bf*)(Ab + (size_t)i * TSTRIDE);
    b0[i] = *(const v16bf*)(Bb + (size_t)i * TSTRIDE);
  }

#pragma unroll 1
  for (int kt = 0; kt < KTILES; kt += 2) {
    // load kt+1 (always valid: KTILES even)
#pragma unroll
    for (int i = 0; i < 4; ++i) {
      a1[i] = *(const v16bf*)(Ab + (size_t)i * TSTRIDE + (size_t)(kt + 1) * TILE_US);
      b1[i] = *(const v16bf*)(Bb + (size_t)i * TSTRIDE + (size_t)(kt + 1) * TILE_US);
    }
    if (kt + 4 < KTILES) {
      __builtin_prefetch(Ab + (size_t)(kt + 4) * TILE_US, 0, 3);
      __builtin_prefetch(Bb + (size_t)(kt + 4) * TILE_US, 0, 3);
    }
#pragma unroll
    for (int i = 0; i < 4; ++i)
#pragma unroll
      for (int j = 0; j < 4; ++j)
        acc[i][j] = __builtin_amdgcn_wmma_f32_16x16x32_bf16(
            false, a0[i], false, b0[j], (short)0, acc[i][j], false, false);
    if (kt + 2 < KTILES) {
#pragma unroll
      for (int i = 0; i < 4; ++i) {
        a0[i] = *(const v16bf*)(Ab + (size_t)i * TSTRIDE + (size_t)(kt + 2) * TILE_US);
        b0[i] = *(const v16bf*)(Bb + (size_t)i * TSTRIDE + (size_t)(kt + 2) * TILE_US);
      }
    }
#pragma unroll
    for (int i = 0; i < 4; ++i)
#pragma unroll
      for (int j = 0; j < 4; ++j)
        acc[i][j] = __builtin_amdgcn_wmma_f32_16x16x32_bf16(
            false, a1[i], false, b1[j], (short)0, acc[i][j], false, false);
  }

  // C layout: vgpr r -> row M = r + 8*(lane>>4); col N = lane&15.
  int lh = lane >> 4, ln = lane & 15;
#pragma unroll
  for (int i = 0; i < 4; ++i) {
    float* o = tmp + (size_t)((mtile + i) * 16 + lh * 8) * NCOLS + ln;
#pragma unroll
    for (int j = 0; j < 4; ++j) {
      size_t c = (size_t)(ntile + j) * 16;
#pragma unroll
      for (int r = 0; r < 8; ++r)
        o[(size_t)r * NCOLS + c] = acc[i][j][r];
    }
  }
}

// ---------- kernel 4: per-step epilogue (one block per batch row) ----------
__global__ void __launch_bounds__(256)
dkt_epilogue(const float* __restrict__ tmp, float* __restrict__ h,
             unsigned short* __restrict__ hA, unsigned short* __restrict__ xA,
             const float* __restrict__ wo, const float* __restrict__ bo,
             const float* __restrict__ b1, const float* __restrict__ b_ih,
             const float* __restrict__ b_hh, const int* __restrict__ prob_ids,
             const int* __restrict__ operate, const float* __restrict__ prob_emb,
             float* __restrict__ out, int t) {
  int b = blockIdx.x, tid = threadIdx.x;
  const float* row = tmp + (size_t)b * NCOLS;
  int pid = prob_ids[t * BDIM + b];
  int j0 = tid * 4;                        // each thread owns 4 consecutive j

  // prob = sigmoid( relu(h@w1+b1) . wo[:,pid] + bo[pid] )
  F4 zr, b1v;
  zr.v  = *(const float4*)(row + j0);
  b1v.v = *(const float4*)(b1 + j0);
  const float* wcol = wo + pid;
  float partial = 0.f;
#pragma unroll
  for (int c = 0; c < 4; ++c) {
    float z = fmaxf(zr.f[c] + b1v.f[c], 0.f);
    partial += z * wcol[(size_t)(j0 + c) * PDIM];
  }
  __shared__ float red[256];
  red[tid] = partial;
  __syncthreads();
  for (int s = 128; s > 0; s >>= 1) {
    if (tid < s) red[tid] += red[tid + s];
    __syncthreads();
  }
  if (tid == 0) out[(size_t)b * TSTEPS + t] = sigmoidf(red[0] + bo[pid]);

  // GRU gates + h update, vectorized by 4
  F4 hr, hz, hn4, ir, iz, in4, bhr, bhz, bhn, bir, biz, bin, hp, ho;
  hr.v  = *(const float4*)(row + 1024 + j0);
  hz.v  = *(const float4*)(row + 2048 + j0);
  hn4.v = *(const float4*)(row + 3072 + j0);
  ir.v  = *(const float4*)(row + 4096 + j0);
  iz.v  = *(const float4*)(row + 5120 + j0);
  in4.v = *(const float4*)(row + 6144 + j0);
  bhr.v = *(const float4*)(b_hh + j0);
  bhz.v = *(const float4*)(b_hh + 1024 + j0);
  bhn.v = *(const float4*)(b_hh + 2048 + j0);
  bir.v = *(const float4*)(b_ih + j0);
  biz.v = *(const float4*)(b_ih + 1024 + j0);
  bin.v = *(const float4*)(b_ih + 2048 + j0);
  hp.v  = *(const float4*)(h + (size_t)b * DIMK + j0);
  unsigned long long pack = 0ull;
#pragma unroll
  for (int c = 0; c < 4; ++c) {
    float r  = sigmoidf((ir.f[c] + bir.f[c]) + (hr.f[c] + bhr.f[c]));
    float u  = sigmoidf((iz.f[c] + biz.f[c]) + (hz.f[c] + bhz.f[c]));
    float n  = tanhf((in4.f[c] + bin.f[c]) + r * (hn4.f[c] + bhn.f[c]));
    float hv = (1.f - u) * n + u * hp.f[c];
    ho.f[c]  = hv;
    pack |= (unsigned long long)f2bf(hv) << (16 * c);
  }
  *(float4*)(h + (size_t)b * DIMK + j0) = ho.v;
  // 4 consecutive kk (kk0 % 4 == 0) are contiguous ushorts in the A-fragment
  {
    size_t base = (size_t)((b >> 4) * KTILES + (j0 >> 5)) * TILE_US +
                  a_frag_off(b & 15, j0 & 31);
    *(unsigned long long*)(hA + base) = pack;
  }

  // gather + pack x for step t+1
  if (t + 1 < TSTEPS) {
    int pid2 = prob_ids[(t + 1) * BDIM + b];
    int op2  = operate[(t + 1) * BDIM + b];
    F4 e; e.v = *(const float4*)(prob_emb + (size_t)(pid2 * 2 + op2) * DIMK + j0);
    unsigned long long px = 0ull;
#pragma unroll
    for (int c = 0; c < 4; ++c) px |= (unsigned long long)f2bf(e.f[c]) << (16 * c);
    size_t base = (size_t)((b >> 4) * KTILES + (j0 >> 5)) * TILE_US +
                  a_frag_off(b & 15, j0 & 31);
    *(unsigned long long*)(xA + base) = px;
  }
}

// ---------- host ----------
extern "C" void kernel_launch(void* const* d_in, const int* in_sizes, int n_in,
                              void* d_out, int out_size, void* d_ws, size_t ws_size,
                              hipStream_t stream) {
  (void)in_sizes; (void)n_in; (void)out_size; (void)ws_size;
  const int*   prob_ids = (const int*)d_in[0];
  const int*   operate  = (const int*)d_in[1];
  // d_in[2] = seq_num (all == T, unused)
  const float* prob_emb = (const float*)d_in[3];
  const float* w1       = (const float*)d_in[4];
  const float* b1       = (const float*)d_in[5];
  const float* wo       = (const float*)d_in[6];
  const float* bo       = (const float*)d_in[7];
  const float* w_ih     = (const float*)d_in[8];
  const float* w_hh     = (const float*)d_in[9];
  const float* b_ih     = (const float*)d_in[10];
  const float* b_hh     = (const float*)d_in[11];
  float* out = (float*)d_out;

  // workspace carve (bytes, 256-aligned); total ~24.1 MB
  char* ws = (char*)d_ws;
  unsigned short* Bp  = (unsigned short*)(ws + 0);         // 14,680,064
  float*          tmp = (float*)(ws + 14680064);           //  7,340,032
  float*          h   = (float*)(ws + 22020096);           //  1,048,576
  unsigned short* hA  = (unsigned short*)(ws + 23068672);  //    524,288
  unsigned short* xA  = (unsigned short*)(ws + 23592960);  //    524,288

  dkt_pack_w<<<(NTILES * KTILES * TILE_US) / 256, 256, 0, stream>>>(w1, w_hh, w_ih, Bp);
  dkt_init<<<(BDIM * DIMK) / 256, 256, 0, stream>>>(prob_ids, operate, prob_emb, h, hA, xA);

  dim3 ggrid(NTILES / 32, MTILES / 4);  // (14, 4)
  for (int t = 0; t < TSTEPS; ++t) {
    dkt_gemm<<<ggrid, 256, 0, stream>>>(hA, xA, Bp, tmp);
    dkt_epilogue<<<BDIM, 256, 0, stream>>>(tmp, h, hA, xA, wo, bo, b1, b_ih, b_hh,
                                           prob_ids, operate, prob_emb, out, t);
  }
}
